// MultiQueryAttention_71734543778305
// MI455X (gfx1250) — matrix operand order repaired and checked by reading, not compiled, not run on hardware
//
#include <hip/hip_runtime.h>
#include <hip/hip_bf16.h>
#include <stdint.h>

// ---------------------------------------------------------------------------
// Types for CDNA5 WMMA (wave32): v_wmma_f32_16x16x32_bf16
// ---------------------------------------------------------------------------
typedef __attribute__((ext_vector_type(16))) __bf16 v16bf;
typedef __attribute__((ext_vector_type(8)))  float  v8f;
typedef __attribute__((ext_vector_type(4)))  unsigned int u32x4;

union FragU { v16bf v; u32x4 q[2]; };

__device__ __forceinline__ __bf16 f2bf(float f) {
  union { float f; unsigned u; } a; a.f = f;
  unsigned r = a.u + 0x7FFFu + ((a.u >> 16) & 1u);  // round-to-nearest-even
  union { unsigned short s; __bf16 b; } o; o.s = (unsigned short)(r >> 16);
  return o.b;
}

// A-matrix 16x32 bf16 fragment (row-major source, contiguous K):
//   lanes 0-15 : row M=lane,     halves 0..7 = K 0..7,  halves 8..15 = K 16..23
//   lanes 16-31: row M=lane-16,  halves 0..7 = K 8..15, halves 8..15 = K 24..31
// rowp points at element (row, k_base); koff = (lane&16)?8:0
__device__ __forceinline__ v16bf ldA(const __bf16* rowp, int koff) {
  FragU f;
  f.q[0] = *reinterpret_cast<const u32x4*>(rowp + koff);
  f.q[1] = *reinterpret_cast<const u32x4*>(rowp + koff + 16);
  return f.v;
}

// B-matrix 32x16 bf16 fragment from column-major storage (Bt[n][k] contiguous):
//   lanes 0-15 : col N=lane,     halves 0..7 = K 0..7,  halves 8..15 = K 16..23
//   lanes 16-31: col N=lane-16,  halves 0..7 = K 8..15, halves 8..15 = K 24..31
// colp points at element (col, k_base); ksel = (lane&16)?16:0
__device__ __forceinline__ v16bf ldB(const __bf16* colp, int ksel) {
  FragU f;
  f.q[0] = *reinterpret_cast<const u32x4*>(colp + (ksel >> 1));
  f.q[1] = *reinterpret_cast<const u32x4*>(colp + (ksel >> 1) + 16);
  return f.v;
}

__device__ __forceinline__ v8f wmma_bf(v16bf a, v16bf b, v8f c) {
  return __builtin_amdgcn_wmma_f32_16x16x32_bf16(false, a, false, b, (short)0, c,
                                                 false, false);
}

// ---------------------------------------------------------------------------
// fp32 -> bf16 conversion / transpose kernels
// ---------------------------------------------------------------------------
__global__ void cvt_bf16_kernel(const float* __restrict__ in,
                                __bf16* __restrict__ out, size_t n) {
  size_t i = (size_t)blockIdx.x * 256 + threadIdx.x;
  if (i < n) out[i] = f2bf(in[i]);
}

// in: [K,N] row-major fp32 -> out: [N,K] row-major bf16
__global__ void transpose_bf16_kernel(const float* __restrict__ in,
                                      __bf16* __restrict__ out, int K, int N) {
  size_t idx = (size_t)blockIdx.x * 256 + threadIdx.x;
  if (idx < (size_t)K * N) {
    int k = (int)(idx / N), n = (int)(idx % N);
    out[(size_t)n * K + k] = f2bf(in[idx]);
  }
}

// ---------------------------------------------------------------------------
// WMMA GEMM: C[M,N] = A[M,K](bf16) * Bt[N,K](bf16)^T, + bias, * scale.
// One wave computes a 16x64 tile: 1 A-frag reused across 4 B-frags
// (4 independent WMMA chains), with register double-buffering of the next
// k-step's fragments so loads overlap the current WMMA group.
// MODE 0: bf16 row-major   MODE 1: bf16 transposed [B,N,seq]   MODE 2: fp32
// blockDim = 128 (4 waves = 4 tiles).
// ---------------------------------------------------------------------------
template <int MODE>
__global__ void __launch_bounds__(128)
gemm_wmma(const __bf16* __restrict__ A, const __bf16* __restrict__ Bt,
          const float* __restrict__ bias, float scale,
          int M, int N, int K, int seq,
          __bf16* __restrict__ outb, float* __restrict__ outf) {
  int lane = threadIdx.x & 31;
  int wave = threadIdx.x >> 5;
  int tile = blockIdx.x * 4 + wave;
  int tiles_n = N >> 6;                       // N / 64
  int tm = tile / tiles_n;
  int tn = tile - tm * tiles_n;
  if (tm * 16 >= M) return;

  int r16  = lane & 15;
  int hi   = (lane >> 4) & 1;
  int koff = hi ? 8 : 0;
  int ksel = hi ? 16 : 0;

  const __bf16* arow = A + (size_t)(tm * 16 + r16) * K;
  const __bf16* brow[4];
#pragma unroll
  for (int c = 0; c < 4; ++c)
    brow[c] = Bt + (size_t)(tn * 64 + c * 16 + r16) * K;

  v8f acc[4];
#pragma unroll
  for (int c = 0; c < 4; ++c) acc[c] = {};

  // prologue loads
  v16bf a_cur = ldA(arow, koff);
  v16bf b_cur[4];
#pragma unroll
  for (int c = 0; c < 4; ++c) b_cur[c] = ldB(brow[c], ksel);

  // main loop: prefetch k+32 fragments, then 4 WMMAs on current fragments
  int k = 0;
  for (; k < K - 32; k += 32) {
    v16bf a_nxt = ldA(arow + k + 32, koff);
    v16bf b_nxt[4];
#pragma unroll
    for (int c = 0; c < 4; ++c) b_nxt[c] = ldB(brow[c] + k + 32, ksel);
#pragma unroll
    for (int c = 0; c < 4; ++c) acc[c] = wmma_bf(a_cur, b_cur[c], acc[c]);
    a_cur = a_nxt;
#pragma unroll
    for (int c = 0; c < 4; ++c) b_cur[c] = b_nxt[c];
  }
  // peeled last k-step (no prefetch -> no OOB reads at buffer tail)
#pragma unroll
  for (int c = 0; c < 4; ++c) acc[c] = wmma_bf(a_cur, b_cur[c], acc[c]);

  // epilogue
#pragma unroll
  for (int c = 0; c < 4; ++c) {
    int col = tn * 64 + c * 16 + r16;
    float bcol = bias ? bias[col] : 0.f;
#pragma unroll
    for (int r = 0; r < 8; ++r) {
      int m = tm * 16 + (hi ? r + 8 : r);
      float v = (acc[c][r] + bcol) * scale;
      if (MODE == 0) {
        outb[(size_t)m * N + col] = f2bf(v);
      } else if (MODE == 1) {
        int bb = m / seq, ss = m - bb * seq;
        outb[((size_t)bb * N + col) * seq + ss] = f2bf(v);
      } else {
        outf[(size_t)m * N + col] = v;
      }
    }
  }
}

// ---------------------------------------------------------------------------
// Flash MQA attention. One wave per (batch, head, 16-query tile).
// Q: [B,S,HID] bf16 (pre-scaled by 1/sqrt(HD)), K: [B,S,HD] bf16,
// Vt: [B,HD,S] bf16, pad: [B,S] fp32, O: [B,S,HID] bf16.
// ---------------------------------------------------------------------------
__global__ void __launch_bounds__(128)
mqa_attn(const __bf16* __restrict__ Q, const __bf16* __restrict__ Kb,
         const __bf16* __restrict__ Vt, const float* __restrict__ pad,
         __bf16* __restrict__ O, int Bn, int S, int NH, int HD) {
  __shared__ __bf16 plds[4][16 * 32];

  int lane = threadIdx.x & 31;
  int wave = threadIdx.x >> 5;
  int gid  = blockIdx.x * 4 + wave;
  int qtiles = S >> 4;
  int qt = gid % qtiles;
  int t  = gid / qtiles;
  int h  = t % NH;
  int b  = t / NH;
  int q0 = qt * 16;
  int HID = NH * HD;

  int r16  = lane & 15;
  int hi   = (lane >> 4) & 1;
  int koff = hi ? 8 : 0;
  int ksel = hi ? 16 : 0;

  // Q tile fragments: 16 rows x 128 (4 K-chunks of 32)
  v16bf qa[4];
#pragma unroll
  for (int kc = 0; kc < 4; ++kc)
    qa[kc] = ldA(Q + (size_t)(b * S + q0 + r16) * HID + h * HD + kc * 32, koff);

  v8f acc[8];
#pragma unroll
  for (int c = 0; c < 8; ++c) acc[c] = {};
  float mi[8], li[8];
#pragma unroll
  for (int r = 0; r < 8; ++r) { mi[r] = -1e30f; li[r] = 0.f; }

  __bf16* myp = plds[wave];
  const __bf16* vbase = Vt + ((size_t)b * HD + r16) * S;

  int jmax = (q0 + 15) >> 5;  // last kv tile with any unmasked key (causal)
  for (int j = 0; j <= jmax; ++j) {
    int k0 = j << 5;
    const __bf16* kr0 = Kb + (size_t)(b * S + k0 + r16) * HD;
    const __bf16* kr1 = kr0 + (size_t)16 * HD;

    // --- load all 8 K-fragments as one clustered batch, then 8 WMMAs ---
    v16bf kf[8];
#pragma unroll
    for (int kc = 0; kc < 4; ++kc) {
      kf[kc]     = ldB(kr0 + kc * 32, ksel);
      kf[kc + 4] = ldB(kr1 + kc * 32, ksel);
    }
    v8f s0 = {}, s1 = {};
#pragma unroll
    for (int kc = 0; kc < 4; ++kc) {
      s0 = wmma_bf(qa[kc], kf[kc], s0);
      s1 = wmma_bf(qa[kc], kf[kc + 4], s1);
    }

    // prefetch next kv tile's K rows + this tile's V rows into cache while
    // the softmax VALU work runs (gfx1250 global_prefetch_b8)
    if (j < jmax) {
      __builtin_prefetch(kr0 + (size_t)32 * HD, 0, 3);
      __builtin_prefetch(kr1 + (size_t)32 * HD, 0, 3);
    }
#pragma unroll
    for (int c = 0; c < 8; ++c)
      __builtin_prefetch(vbase + (size_t)c * 16 * S + k0, 0, 3);

    int ki0 = k0 + r16, ki1 = ki0 + 16;
    float pm0 = pad[(size_t)b * S + ki0] * -1e9f;
    float pm1 = pad[(size_t)b * S + ki1] * -1e9f;

    // --- online softmax, per lane covering 8 rows ---
#pragma unroll
    for (int r = 0; r < 8; ++r) {
      int rowm = hi ? r + 8 : r;
      int qi = q0 + rowm;
      float v0 = s0[r] + pm0 + (ki0 > qi ? -1e9f : 0.f);
      float v1 = s1[r] + pm1 + (ki1 > qi ? -1e9f : 0.f);
      float mx = fmaxf(v0, v1);
#pragma unroll
      for (int msk = 1; msk < 16; msk <<= 1) mx = fmaxf(mx, __shfl_xor(mx, msk, 32));
      float mnew = fmaxf(mi[r], mx);
      float corr = __expf(mi[r] - mnew);
      mi[r] = mnew;
      float p0 = __expf(v0 - mnew);
      float p1 = __expf(v1 - mnew);
      float ps = p0 + p1;
#pragma unroll
      for (int msk = 1; msk < 16; msk <<= 1) ps += __shfl_xor(ps, msk, 32);
      li[r] = li[r] * corr + ps;
#pragma unroll
      for (int c = 0; c < 8; ++c) acc[c][r] *= corr;
      // D-layout -> LDS row-major 16x32 tile (transpose staging for A-frag)
      myp[rowm * 32 + r16]      = f2bf(p0);
      myp[rowm * 32 + r16 + 16] = f2bf(p1);
    }

    // --- P fragment (16x32) from LDS, V frags batched, O += P @ V ---
    v16bf pf = ldA(myp + r16 * 32, koff);
    v16bf vf[8];
#pragma unroll
    for (int c = 0; c < 8; ++c)
      vf[c] = ldB(vbase + (size_t)c * 16 * S + k0, ksel);
#pragma unroll
    for (int c = 0; c < 8; ++c)
      acc[c] = wmma_bf(pf, vf[c], acc[c]);
  }

  // --- epilogue: O /= l, store bf16 [B,S,HID] at this head's slice ---
#pragma unroll
  for (int r = 0; r < 8; ++r) {
    int rowm = hi ? r + 8 : r;
    float inv = 1.f / li[r];
    size_t base = (size_t)(b * S + q0 + rowm) * HID + h * HD;
#pragma unroll
    for (int c = 0; c < 8; ++c)
      O[base + c * 16 + r16] = f2bf(acc[c][r] * inv);
  }
}

// ---------------------------------------------------------------------------
// Host launcher
// ---------------------------------------------------------------------------
extern "C" void kernel_launch(void* const* d_in, const int* in_sizes, int n_in,
                              void* d_out, int out_size, void* d_ws, size_t ws_size,
                              hipStream_t stream) {
  (void)in_sizes; (void)n_in; (void)out_size; (void)ws_size;

  constexpr int Bn = 2, S = 2048, HID = 2048, NH = 16, HD = 128;
  const size_t M = (size_t)Bn * S;  // 4096 token rows

  const float* hidden = (const float*)d_in[0];
  // d_in[1] = causal mask (recomputed analytically, unused)
  const float* pad = (const float*)d_in[2];
  const float* Wq = (const float*)d_in[3];
  const float* bq = (const float*)d_in[4];
  const float* Wk = (const float*)d_in[5];
  const float* bk = (const float*)d_in[6];
  const float* Wv = (const float*)d_in[7];
  const float* bv = (const float*)d_in[8];
  const float* Wo = (const float*)d_in[9];
  const float* bo = (const float*)d_in[10];
  float* out = (float*)d_out;

  // workspace carve-up (bf16 buffers), 256B aligned
  char* ws = (char*)d_ws;
  auto carve = [&](size_t bytes) {
    char* p = ws;
    ws += (bytes + 255) & ~(size_t)255;
    return p;
  };
  __bf16* hbf = (__bf16*)carve(M * HID * 2);             // hidden bf16
  __bf16* wqt = (__bf16*)carve((size_t)HID * HID * 2);   // Wq^T [HID,HID]
  __bf16* wkt = (__bf16*)carve((size_t)HD * HID * 2);    // Wk^T [HD,HID]
  __bf16* wvt = (__bf16*)carve((size_t)HD * HID * 2);    // Wv^T [HD,HID]
  __bf16* wot = (__bf16*)carve((size_t)HID * HID * 2);   // Wo^T [HID,HID]
  __bf16* qbf = (__bf16*)carve(M * HID * 2);             // Q [B,S,HID]
  __bf16* kbf = (__bf16*)carve(M * HD * 2);              // K [B,S,HD]
  __bf16* vtb = (__bf16*)carve((size_t)Bn * HD * S * 2); // V^T [B,HD,S]
  __bf16* obf = (__bf16*)carve(M * HID * 2);             // attn out [B,S,HID]

  const float qscale = 0.088388347648318447f;  // 1/sqrt(128)

  // 1) conversions / weight transposes
  cvt_bf16_kernel<<<(unsigned)((M * HID + 255) / 256), 256, 0, stream>>>(hidden, hbf, M * HID);
  transpose_bf16_kernel<<<(unsigned)(((size_t)HID * HID + 255) / 256), 256, 0, stream>>>(Wq, wqt, HID, HID);
  transpose_bf16_kernel<<<(unsigned)(((size_t)HID * HD + 255) / 256), 256, 0, stream>>>(Wk, wkt, HID, HD);
  transpose_bf16_kernel<<<(unsigned)(((size_t)HID * HD + 255) / 256), 256, 0, stream>>>(Wv, wvt, HID, HD);
  transpose_bf16_kernel<<<(unsigned)(((size_t)HID * HID + 255) / 256), 256, 0, stream>>>(Wo, wot, HID, HID);

  // 2) projections (WMMA GEMMs); 16x64 tile per wave, 4 waves per block
  unsigned q_blocks  = (unsigned)((M / 16) * (HID / 64) / 4);  // 2048
  unsigned kv_blocks = (unsigned)((M / 16) * (HD / 64) / 4);   // 128
  gemm_wmma<0><<<q_blocks, 128, 0, stream>>>(hbf, wqt, bq, qscale, (int)M, HID, HID, S, qbf, nullptr);
  gemm_wmma<0><<<kv_blocks, 128, 0, stream>>>(hbf, wkt, bk, 1.f, (int)M, HD, HID, S, kbf, nullptr);
  gemm_wmma<1><<<kv_blocks, 128, 0, stream>>>(hbf, wvt, bv, 1.f, (int)M, HD, HID, S, vtb, nullptr);

  // 3) flash MQA attention: one wave per (b, h, 16-query tile)
  unsigned attn_blocks = (unsigned)((size_t)Bn * NH * (S / 16) / 4);  // 1024
  mqa_attn<<<attn_blocks, 128, 0, stream>>>(qbf, kbf, vtb, pad, obf, Bn, S, NH, HD);

  // 4) output projection -> fp32 d_out
  gemm_wmma<2><<<q_blocks, 128, 0, stream>>>(obf, wot, bo, 1.f, (int)M, HID, HID, S, nullptr, out);
}
